// QAttention_32195074850894
// MI455X (gfx1250) — matrix operand order repaired and checked by reading, not compiled, not run on hardware
//
#include <hip/hip_runtime.h>

// ---------------------------------------------------------------------------
// QAttention (PVT spatial-reduction attention with fake-quant) for gfx1250.
// All GEMMs on v_wmma_f32_16x16x32_f16 (wave32, 16x16 tile per wave).
// q/k/v snapped to their int8 grids once (after scale reduction), so the hot
// attention loops are pure b128/ds_load + WMMA. K and V^T staged in LDS.
// ---------------------------------------------------------------------------

typedef __attribute__((ext_vector_type(16))) _Float16 v16h;
typedef __attribute__((ext_vector_type(8)))  _Float16 v8h;
typedef __attribute__((ext_vector_type(8)))  float    v8f;

#define BSZ   8
#define NSEQ  4096
#define CDIM  256
#define HEADS 8
#define DHEAD 32
#define N2    256   // (64/4)*(64/4)
#define KSR   4096  // C * 4 * 4

__device__ __forceinline__ int lane_id() { return threadIdx.x & 31; }
__device__ __forceinline__ int wave_id() { return threadIdx.x >> 5; }

// symmetric fake-quant snap: clip(round(x/d), -128, 127) * d   (RNE == jnp.round)
__device__ __forceinline__ float fq_sym(float x, float d) {
    float r = rintf(x / d);
    r = fminf(fmaxf(r, -128.f), 127.f);
    return r * d;
}

__device__ __forceinline__ v8f wmma_f16(v16h a, v16h b, v8f c) {
    return __builtin_amdgcn_wmma_f32_16x16x32_f16(false, a, false, b, (short)0, c,
                                                  false, false);
}

// A-fragment: 16x32 f16 (row-major, leading dim lda).
// ISA layout: lanes 0-15 hold K=0..7 & 16..23 of row (lane&15); lanes 16-31 hold
// K=8..15 & 24..31.
__device__ __forceinline__ v16h load_a(const _Float16* A, int lda) {
    int L = lane_id();
    const _Float16* p = A + (L & 15) * lda + ((L < 16) ? 0 : 8);
    v8h lo = *(const v8h*)p;
    v8h hi = *(const v8h*)(p + 16);
    v16h r;
#pragma unroll
    for (int i = 0; i < 8; ++i) { r[i] = lo[i]; r[8 + i] = hi[i]; }
    return r;
}

// B-fragment from W stored row-major [N,K] (i.e. B = W^T): lane L holds column
// n=(L&15) of B, K=0..7 & 16..23 (lanes 0-15) / K=8..15 & 24..31 (lanes 16-31)
// (two 16B loads, matching load_a).
__device__ __forceinline__ v16h load_bt(const _Float16* W, int ldw) {
    int L = lane_id();
    const _Float16* p = W + (L & 15) * ldw + ((L < 16) ? 0 : 8);
    v8h lo = *(const v8h*)p;
    v8h hi = *(const v8h*)(p + 16);
    v16h r;
#pragma unroll
    for (int i = 0; i < 8; ++i) { r[i] = lo[i]; r[8 + i] = hi[i]; }
    return r;
}

// ---------------------------------------------------------------------------
// scale slots in ws (uint bit patterns):
// 0 qw 1 kvw 2 srw 3 projw 4 q 5 k 6 v 7 attn_min 8 attn_max
// ---------------------------------------------------------------------------
__global__ void k_init(unsigned* s) {
    if (threadIdx.x < 16) s[threadIdx.x] = (threadIdx.x == 7) ? 0x7F800000u : 0u;
}

__global__ void k_maxabs(const float* __restrict__ x, int n, unsigned* slot) {
    float m = 0.f;
    for (int i = blockIdx.x * blockDim.x + threadIdx.x; i < n;
         i += gridDim.x * blockDim.x)
        m = fmaxf(m, fabsf(x[i]));
#pragma unroll
    for (int o = 16; o > 0; o >>= 1) m = fmaxf(m, __shfl_xor(m, o, 32));
    if (lane_id() == 0) atomicMax(slot, __float_as_uint(m));
}

__global__ void k_quant_w(const float* __restrict__ w, int n,
                          const unsigned* __restrict__ slot, _Float16* out) {
    float d = __uint_as_float(*slot) / 127.f + 1e-8f;
    int i = blockIdx.x * blockDim.x + threadIdx.x;
    if (i < n) out[i] = (_Float16)fq_sym(w[i], d);
}

__global__ void k_cvt_h(const float* __restrict__ x, int n, _Float16* out) {
    int i = blockIdx.x * blockDim.x + threadIdx.x;
    if (i < n) out[i] = (_Float16)x[i];
}

// snap an f16 activation buffer to its int8 grid in place
__global__ void k_quant_h(_Float16* x, int n, const unsigned* __restrict__ slot) {
    float d = __uint_as_float(*slot) / 127.f + 1e-8f;
    int i = blockIdx.x * blockDim.x + threadIdx.x;
    if (i < n) x[i] = (_Float16)fq_sym((float)x[i], d);
}

// quantize v and transpose per (b,h): vh[bh, m2, d] -> vt[bh, d, m2]
__global__ void k_quant_vt(const _Float16* __restrict__ vh, _Float16* __restrict__ vt,
                           const unsigned* __restrict__ slot) {
    float d = __uint_as_float(*slot) / 127.f + 1e-8f;
    int i = blockIdx.x * blockDim.x + threadIdx.x;  // 524288
    int bh = i >> 13, rem = i & 8191, m2 = rem >> 5, dd = rem & 31;
    vt[((size_t)bh * DHEAD + dd) * N2 + m2] = (_Float16)fq_sym((float)vh[i], d);
}

// ---------------------------------------------------------------------------
// q = x @ q_w^T + q_b ; write permuted [B,heads,N,dh] f16, reduce max|q|
// ---------------------------------------------------------------------------
__global__ void k_gemm_q(const _Float16* __restrict__ xh,
                         const _Float16* __restrict__ wq,
                         const float* __restrict__ bias, _Float16* __restrict__ qh,
                         unsigned* slotq) {
    int w = blockIdx.x * 8 + wave_id();
    int tm = w >> 4, tn = w & 15;
    const _Float16* A = xh + (size_t)tm * 16 * CDIM;
    const _Float16* W = wq + (size_t)tn * 16 * CDIM;
    v8f acc = {};
    for (int k = 0; k < CDIM; k += 32) {
        __builtin_prefetch(A + k + 128, 0, 1);
        v16h a = load_a(A + k, CDIM);
        v16h b = load_bt(W + k, CDIM);
        acc = wmma_f16(a, b, acc);
    }
    int L = lane_id();
    int col = tn * 16 + (L & 15);
    int g = (L >> 4) << 3;
    int h = col >> 5, dd = col & 31;
    float bs = bias[col];
    float mx = 0.f;
#pragma unroll
    for (int i = 0; i < 8; ++i) {
        int m = tm * 16 + g + i;
        float v = acc[i] + bs;
        mx = fmaxf(mx, fabsf(v));
        int b_ = m >> 12, n_ = m & 4095;
        qh[(size_t)(((b_ * HEADS + h) * NSEQ) + n_) * DHEAD + dd] = (_Float16)v;
    }
#pragma unroll
    for (int o = 16; o > 0; o >>= 1) mx = fmaxf(mx, __shfl_xor(mx, o, 32));
    if (L == 0) atomicMax(slotq, __float_as_uint(mx));
}

// ---------------------------------------------------------------------------
// spatial-reduction conv as gather-GEMM: xc[b*256+n2, co] (f32, +sr_b)
// ---------------------------------------------------------------------------
__global__ void k_gemm_sr(const _Float16* __restrict__ xh,
                          const _Float16* __restrict__ srw,
                          const float* __restrict__ bias, float* __restrict__ xc) {
    int w = blockIdx.x * 8 + wave_id();
    int tm = w >> 4, tn = w & 15;  // 128 x 16 tiles
    int L = lane_id();
    int ml = L & 15;
    int kb = (L < 16) ? 0 : 8;
    int mg = tm * 16 + ml;
    int b_ = mg >> 8, n2 = mg & 255;
    int oh = n2 >> 4, ow = n2 & 15;
    const _Float16* xb = xh + (size_t)b_ * NSEQ * CDIM;
    const _Float16* W = srw + (size_t)tn * 16 * KSR;
    v8f acc = {};
    for (int k = 0; k < KSR; k += 32) {
        v16h a;
#pragma unroll
        for (int i = 0; i < 16; ++i) {
            int kk = k + ((i < 8) ? (kb + i) : (16 + kb + (i - 8)));
            int ci = kk >> 4, khh = (kk >> 2) & 3, kww = kk & 3;
            int row = (oh * 4 + khh) * 64 + (ow * 4 + kww);
            a[i] = xb[(size_t)row * CDIM + ci];
        }
        v16h b = load_bt(W + k, KSR);
        acc = wmma_f16(a, b, acc);
    }
    int col = tn * 16 + (L & 15);
    int g = (L >> 4) << 3;
    float bs = bias[col];
#pragma unroll
    for (int i = 0; i < 8; ++i)
        xc[(size_t)(tm * 16 + g + i) * CDIM + col] = acc[i] + bs;
}

// LayerNorm over C=256, one wave per row -> f16
__global__ void k_ln(const float* __restrict__ xc, const float* __restrict__ gam,
                     const float* __restrict__ bet, _Float16* __restrict__ xnh) {
    int row = blockIdx.x * 8 + wave_id();
    int L = lane_id();
    const float* p = xc + (size_t)row * CDIM;
    float vals[8], s = 0.f, s2 = 0.f;
#pragma unroll
    for (int i = 0; i < 8; ++i) {
        float v = p[L * 8 + i];
        vals[i] = v; s += v; s2 += v * v;
    }
#pragma unroll
    for (int o = 16; o > 0; o >>= 1) {
        s += __shfl_xor(s, o, 32);
        s2 += __shfl_xor(s2, o, 32);
    }
    float mu = s * (1.f / CDIM);
    float var = s2 * (1.f / CDIM) - mu * mu;
    float inv = rsqrtf(var + 1e-5f);
#pragma unroll
    for (int i = 0; i < 8; ++i) {
        int c = L * 8 + i;
        xnh[(size_t)row * CDIM + c] = (_Float16)((vals[i] - mu) * inv * gam[c] + bet[c]);
    }
}

// kv = xn @ kv_w^T + kv_b -> kh/vh [B,heads,N2,dh] f16, max|k|,max|v|
__global__ void k_gemm_kv(const _Float16* __restrict__ xnh,
                          const _Float16* __restrict__ wkv,
                          const float* __restrict__ bias, _Float16* __restrict__ kh,
                          _Float16* __restrict__ vh, unsigned* slotk, unsigned* slotv) {
    int w = blockIdx.x * 8 + wave_id();
    int tm = w >> 5, tn = w & 31;
    const _Float16* A = xnh + (size_t)tm * 16 * CDIM;
    const _Float16* W = wkv + (size_t)tn * 16 * CDIM;
    v8f acc = {};
    for (int k = 0; k < CDIM; k += 32) {
        v16h a = load_a(A + k, CDIM);
        v16h b = load_bt(W + k, CDIM);
        acc = wmma_f16(a, b, acc);
    }
    int L = lane_id();
    int col = tn * 16 + (L & 15);  // [0,512)
    int g = (L >> 4) << 3;
    int sv = col >> 8;             // 0=k 1=v (tile-uniform)
    int h = (col >> 5) & 7, dd = col & 31;
    _Float16* dst = sv ? vh : kh;
    float bs = bias[col];
    float mx = 0.f;
#pragma unroll
    for (int i = 0; i < 8; ++i) {
        int m = tm * 16 + g + i;
        float v = acc[i] + bs;
        mx = fmaxf(mx, fabsf(v));
        int b_ = m >> 8, n2 = m & 255;
        dst[(size_t)(((b_ * HEADS + h) * N2) + n2) * DHEAD + dd] = (_Float16)v;
    }
#pragma unroll
    for (int o = 16; o > 0; o >>= 1) mx = fmaxf(mx, __shfl_xor(mx, o, 32));
    if (L == 0) atomicMax(sv ? slotv : slotk, __float_as_uint(mx));
}

// logits (16 rows x 256 cols) via 16 WMMAs (K=dh=32); scaled softmax in regs.
// q/k are pre-snapped to their int8 grids; K tile comes from LDS.
__device__ __forceinline__ void attn_softmax(const _Float16* Q, const _Float16* Ksh,
                                             v8f acc[16], float rsum[8]) {
    v16h a = load_a(Q, DHEAD);
#pragma unroll
    for (int t = 0; t < 16; ++t) {
        v16h b = load_bt(Ksh + t * 16 * DHEAD, DHEAD);
        v8f z = {};
        acc[t] = wmma_f16(a, b, z);
    }
    const float scl = 0.17677669529663687f;  // 1/sqrt(32)
#pragma unroll
    for (int t = 0; t < 16; ++t)
#pragma unroll
        for (int i = 0; i < 8; ++i) acc[t][i] *= scl;
    float rmax[8];
#pragma unroll
    for (int i = 0; i < 8; ++i) {
        float m = -3.0e38f;
#pragma unroll
        for (int t = 0; t < 16; ++t) m = fmaxf(m, acc[t][i]);
#pragma unroll
        for (int o = 8; o > 0; o >>= 1) m = fmaxf(m, __shfl_xor(m, o, 32));
        rmax[i] = m;
    }
#pragma unroll
    for (int i = 0; i < 8; ++i) {
        float s = 0.f;
#pragma unroll
        for (int t = 0; t < 16; ++t) {
            float e = __expf(acc[t][i] - rmax[i]);
            acc[t][i] = e; s += e;
        }
#pragma unroll
        for (int o = 8; o > 0; o >>= 1) s += __shfl_xor(s, o, 32);
        rsum[i] = 1.f / s;
    }
}

// pass 1: global min/max of softmax probs (no store; recompute in pass 2).
// All 8 waves of a block share (b,h): stage K tile (16 KB) in LDS once.
__global__ void k_attn_mm(const _Float16* __restrict__ qh,
                          const _Float16* __restrict__ kh, unsigned* mm) {
    __shared__ _Float16 ksh[N2 * DHEAD];  // 16 KB
    int w = blockIdx.x * 8 + wave_id();
    int bh = w >> 8, rt = w & 255;
    {
        const _Float16* Kg = kh + (size_t)bh * N2 * DHEAD;
        for (int i = threadIdx.x; i < (N2 * DHEAD) / 8; i += blockDim.x)
            ((v8h*)ksh)[i] = ((const v8h*)Kg)[i];
    }
    __syncthreads();
    v8f acc[16]; float rs[8];
    attn_softmax(qh + (size_t)(bh * NSEQ + rt * 16) * DHEAD, ksh, acc, rs);
    float mn = 3.0e38f, mx = 0.f;
#pragma unroll
    for (int t = 0; t < 16; ++t)
#pragma unroll
        for (int i = 0; i < 8; ++i) {
            float p = acc[t][i] * rs[i];
            mn = fminf(mn, p); mx = fmaxf(mx, p);
        }
#pragma unroll
    for (int o = 16; o > 0; o >>= 1) {
        mn = fminf(mn, __shfl_xor(mn, o, 32));
        mx = fmaxf(mx, __shfl_xor(mx, o, 32));
    }
    if (lane_id() == 0) {
        atomicMin(&mm[0], __float_as_uint(mn));
        atomicMax(&mm[1], __float_as_uint(mx));
    }
}

// pass 2: recompute softmax, asym-quant probs through LDS, attn @ v on WMMA.
// 4 waves/block share (b,h): K (16K) + V^T (16K) staged in LDS; prob tiles 32K.
__global__ __launch_bounds__(128) void k_attn_out(
    const _Float16* __restrict__ qh, const _Float16* __restrict__ kh,
    const _Float16* __restrict__ vt, const unsigned* __restrict__ sl,
    _Float16* __restrict__ outh) {
    __shared__ _Float16 ksh[N2 * DHEAD];      // 16 KB
    __shared__ _Float16 vsh[DHEAD * N2];      // 16 KB, [d, m2]
    __shared__ _Float16 pt[4 * 16 * 256];     // 32 KB, one 16x256 tile per wave
    int wv = wave_id();
    int w = blockIdx.x * 4 + wv;
    int bh = w >> 8, rt = w & 255;
    int b_ = bh >> 3, h = bh & 7;
    {
        const _Float16* Kg = kh + (size_t)bh * N2 * DHEAD;
        const _Float16* Vg = vt + (size_t)bh * DHEAD * N2;
        for (int i = threadIdx.x; i < (N2 * DHEAD) / 8; i += blockDim.x) {
            ((v8h*)ksh)[i] = ((const v8h*)Kg)[i];
            ((v8h*)vsh)[i] = ((const v8h*)Vg)[i];
        }
    }
    __syncthreads();
    v8f acc[16]; float rs[8];
    attn_softmax(qh + (size_t)(bh * NSEQ + rt * 16) * DHEAD, ksh, acc, rs);
    float amin = __uint_as_float(sl[7]);
    float amax = __uint_as_float(sl[8]);
    float da = (amax - amin) / 255.f + 1e-8f;
    float zp = rintf(-amin / da);
    _Float16* tile = pt + wv * 16 * 256;
    int L = lane_id();
    int g = (L >> 4) << 3;
    int cl = L & 15;
#pragma unroll
    for (int t = 0; t < 16; ++t)
#pragma unroll
        for (int i = 0; i < 8; ++i) {
            float p = acc[t][i] * rs[i];
            float qv = fminf(fmaxf(rintf(p / da) + zp, 0.f), 255.f);
            tile[(g + i) * 256 + t * 16 + cl] = (_Float16)((qv - zp) * da);
        }
    __syncthreads();
    v8f o0 = {}, o1 = {};
    for (int k = 0; k < N2; k += 32) {
        v16h a = load_a(tile + k, 256);
        v16h b0 = load_bt(vsh + k, 256);            // d-rows 0..15
        v16h b1 = load_bt(vsh + 16 * 256 + k, 256); // d-rows 16..31
        o0 = wmma_f16(a, b0, o0);
        o1 = wmma_f16(a, b1, o1);
    }
#pragma unroll
    for (int i = 0; i < 8; ++i) {
        int n_ = rt * 16 + g + i;
        size_t base = ((size_t)b_ * NSEQ + n_) * CDIM + h * DHEAD;
        outh[base + cl] = (_Float16)o0[i];
        outh[base + 16 + cl] = (_Float16)o1[i];
    }
}

// final projection: out = attn_out @ proj_w^T + proj_b (f32 to d_out)
__global__ void k_gemm_proj(const _Float16* __restrict__ oh,
                            const _Float16* __restrict__ wp,
                            const float* __restrict__ bias, float* __restrict__ out) {
    int w = blockIdx.x * 8 + wave_id();
    int tm = w >> 4, tn = w & 15;
    const _Float16* A = oh + (size_t)tm * 16 * CDIM;
    const _Float16* W = wp + (size_t)tn * 16 * CDIM;
    v8f acc = {};
    for (int k = 0; k < CDIM; k += 32) {
        __builtin_prefetch(A + k + 128, 0, 1);
        v16h a = load_a(A + k, CDIM);
        v16h b = load_bt(W + k, CDIM);
        acc = wmma_f16(a, b, acc);
    }
    int L = lane_id();
    int col = tn * 16 + (L & 15);
    int g = (L >> 4) << 3;
    float bs = bias[col];
#pragma unroll
    for (int i = 0; i < 8; ++i)
        out[(size_t)(tm * 16 + g + i) * CDIM + col] = acc[i] + bs;
}

// ---------------------------------------------------------------------------
extern "C" void kernel_launch(void* const* d_in, const int* in_sizes, int n_in,
                              void* d_out, int out_size, void* d_ws, size_t ws_size,
                              hipStream_t stream) {
    const float* x      = (const float*)d_in[0];
    const float* q_w    = (const float*)d_in[1];
    const float* q_b    = (const float*)d_in[2];
    const float* kv_w   = (const float*)d_in[3];
    const float* kv_b   = (const float*)d_in[4];
    const float* sr_w   = (const float*)d_in[5];
    const float* sr_b   = (const float*)d_in[6];
    const float* norm_g = (const float*)d_in[7];
    const float* norm_b = (const float*)d_in[8];
    const float* proj_w = (const float*)d_in[9];
    const float* proj_b = (const float*)d_in[10];

    char* ws = (char*)d_ws;
    unsigned* scal = (unsigned*)ws;
    size_t off = 256;
    auto alloc = [&](size_t bytes) {
        void* p = ws + off;
        off += (bytes + 255) & ~(size_t)255;
        return p;
    };
    _Float16* xh   = (_Float16*)alloc((size_t)BSZ * NSEQ * CDIM * 2);   // 16.8 MB
    _Float16* qwh  = (_Float16*)alloc((size_t)CDIM * CDIM * 2);
    _Float16* kvwh = (_Float16*)alloc((size_t)2 * CDIM * CDIM * 2);
    _Float16* srwh = (_Float16*)alloc((size_t)CDIM * KSR * 2);
    _Float16* pjwh = (_Float16*)alloc((size_t)CDIM * CDIM * 2);
    _Float16* qh   = (_Float16*)alloc((size_t)BSZ * NSEQ * CDIM * 2);   // 16.8 MB
    float*    xc   = (float*)alloc((size_t)BSZ * N2 * CDIM * 4);
    _Float16* xnh  = (_Float16*)alloc((size_t)BSZ * N2 * CDIM * 2);
    _Float16* kh   = (_Float16*)alloc((size_t)BSZ * HEADS * N2 * DHEAD * 2);
    _Float16* vh   = (_Float16*)alloc((size_t)BSZ * HEADS * N2 * DHEAD * 2);
    _Float16* vtq  = (_Float16*)alloc((size_t)BSZ * HEADS * DHEAD * N2 * 2);
    _Float16* outh = (_Float16*)alloc((size_t)BSZ * NSEQ * CDIM * 2);   // 16.8 MB

    k_init<<<1, 32, 0, stream>>>(scal);

    k_maxabs<<<64, 256, 0, stream>>>(q_w, CDIM * CDIM, scal + 0);
    k_maxabs<<<128, 256, 0, stream>>>(kv_w, 2 * CDIM * CDIM, scal + 1);
    k_maxabs<<<512, 256, 0, stream>>>(sr_w, CDIM * KSR, scal + 2);
    k_maxabs<<<64, 256, 0, stream>>>(proj_w, CDIM * CDIM, scal + 3);

    k_quant_w<<<256, 256, 0, stream>>>(q_w, CDIM * CDIM, scal + 0, qwh);
    k_quant_w<<<512, 256, 0, stream>>>(kv_w, 2 * CDIM * CDIM, scal + 1, kvwh);
    k_quant_w<<<4096, 256, 0, stream>>>(sr_w, CDIM * KSR, scal + 2, srwh);
    k_quant_w<<<256, 256, 0, stream>>>(proj_w, CDIM * CDIM, scal + 3, pjwh);

    k_cvt_h<<<32768, 256, 0, stream>>>(x, BSZ * NSEQ * CDIM, xh);

    k_gemm_q<<<4096, 256, 0, stream>>>(xh, qwh, q_b, qh, scal + 4);
    k_gemm_sr<<<256, 256, 0, stream>>>(xh, srwh, sr_b, xc);
    k_ln<<<256, 256, 0, stream>>>(xc, norm_g, norm_b, xnh);
    k_gemm_kv<<<512, 256, 0, stream>>>(xnh, kvwh, kv_b, kh, vh, scal + 5, scal + 6);

    // snap q/k to int8 grid in place; build quantized V^T [bh, d, m2]
    k_quant_h<<<32768, 256, 0, stream>>>(qh, BSZ * NSEQ * CDIM, scal + 4);
    k_quant_h<<<2048, 256, 0, stream>>>(kh, BSZ * HEADS * N2 * DHEAD, scal + 5);
    k_quant_vt<<<2048, 256, 0, stream>>>(vh, vtq, scal + 6);

    k_attn_mm<<<2048, 256, 0, stream>>>(qh, kh, scal + 7);
    k_attn_out<<<4096, 128, 0, stream>>>(qh, kh, vtq, scal, outh);

    k_gemm_proj<<<4096, 256, 0, stream>>>(outh, pjwh, proj_b, (float*)d_out);

    (void)in_sizes; (void)n_in; (void)out_size; (void)ws_size;
}
